// PVConvBlock_79456894976546
// MI455X (gfx1250) — matrix operand order repaired and checked by reading, not compiled, not run on hardware
//
#include <hip/hip_runtime.h>
#include <hip/hip_bf16.h>
#include <math.h>

typedef __attribute__((ext_vector_type(16))) __bf16 v16bf;
typedef __attribute__((ext_vector_type(8)))  float  v8f;

#define RV   32
#define R3   32768
#define BB   4
#define CIN  32
#define CO   64
#define NP   32768
#define TDIM 256
#define NGR  8
#define CPG  8
#define WPE  110592   // 4 oc-tiles * 27 taps * 2 kc * 512 (lane*16+e) prepped weights

__device__ __forceinline__ float silu_f(float x) {
    return x / (1.0f + __expf(-x));
}

// ---------------------------------------------------------------------------
// Pointwise GEMM: Y[b,o,n] = sum_c W[o,c] * X[b,c,n]    (O=64 fixed, Cdim param)
// One wave per 16(out-ch) x 16(point) tile, K-chunks of 32 via WMMA bf16.
// tiles = B * (CO/16) * (NP/16) = 4*4*2048 = 32768; grid 4096 x 256 threads.
// ---------------------------------------------------------------------------
__global__ void k_gemm_wmma(const float* __restrict__ W, const float* __restrict__ X,
                            float* __restrict__ Y, int Cdim) {
    const int lane = threadIdx.x & 31;
    const int wave = threadIdx.x >> 5;
    const int tile = blockIdx.x * 8 + wave;
    const int nt  = tile & 2047;          // point tile
    const int mo  = (tile >> 11) & 3;     // out-channel tile
    const int b   = tile >> 13;           // batch
    const int row = lane & 15;
    const int h   = lane >> 4;
    const int col = row;

    v8f acc = {};
    const int kchunks = Cdim >> 5;
    for (int kc = 0; kc < kchunks; ++kc) {
        const int kbase = kc << 5;
        v16bf a, bm;
        #pragma unroll
        for (int e = 0; e < 16; ++e) {
            const int Ka = (e < 8) ? (h * 8 + e) : (16 + h * 8 + (e - 8));
            a[e] = (__bf16)W[(mo * 16 + row) * Cdim + kbase + Ka];
            const int Kb = (e < 8) ? (h * 8 + e) : (16 + h * 8 + (e - 8));
            bm[e] = (__bf16)X[(size_t)(b * Cdim + kbase + Kb) * NP + nt * 16 + col];
        }
        acc = __builtin_amdgcn_wmma_f32_16x16x32_bf16(false, a, false, bm,
                                                      (short)0, acc, false, false);
    }
    #pragma unroll
    for (int r = 0; r < 8; ++r) {
        const int m = r + h * 8;
        Y[(size_t)(b * CO + mo * 16 + m) * NP + nt * 16 + col] = acc[r];
    }
}

// ---------------------------------------------------------------------------
// Pre-swizzle conv weights (O,I,3,3,3) f32 -> bf16 in WMMA A-fragment order:
// wp[((mo*27+tap)*2+kc)*512 + lane*16 + e], so each lane reads 32 contiguous B.
// ---------------------------------------------------------------------------
__global__ void k_prep_wconv(const float* __restrict__ w, __bf16* __restrict__ wp) {
    const int i = blockIdx.x * 256 + threadIdx.x;
    if (i >= WPE) return;
    const int e    = i & 15;
    const int lane = (i >> 4) & 31;
    const int kc   = (i >> 9) & 1;
    const int t2   = i >> 10;            // mo*27 + tap
    const int tap  = t2 % 27;
    const int mo   = t2 / 27;
    const int row  = lane & 15;
    const int h    = lane >> 4;
    const int Ka   = h * 16 + e;
    wp[i] = (__bf16)w[(mo * 16 + row) * 1728 + (kc * 32 + Ka) * 27 + tap];
}

// ---------------------------------------------------------------------------
// 3x3x3 SAME conv, 64->64 ch on 32^3 grid, implicit GEMM via WMMA + LDS halo.
// Block (256 thr, 8 waves) owns (b, x, 2 y-rows). Halo: 64ch x 3x x 4y x 32z
// staged once in LDS as bf16, channel-minor -> B fragments are contiguous.
// grid = B * 32 * 16 = 2048 blocks. 16 wmma-tiles/block, 2 per wave.
// ---------------------------------------------------------------------------
__global__ void k_conv3d_wmma(const float* __restrict__ in, const __bf16* __restrict__ wp,
                              float* __restrict__ out) {
    __shared__ __bf16 lds[384 * 64];     // 49,152 B

    const int tid  = threadIdx.x;
    const int lane = tid & 31;
    const int wave = tid >> 5;
    const int bid  = blockIdx.x;
    const int yt   = bid & 15;
    const int xx   = (bid >> 4) & 31;
    const int b    = bid >> 9;
    const int row  = lane & 15;
    const int h    = lane >> 4;
    const int col  = row;

    // ---- stage halo: pos p = (xo*4 + yo)*32 + z, channel-minor bf16 ----
    const float* inb = in + (size_t)b * CO * R3;
    for (int p = tid; p < 384; p += 256) {
        const int xo  = p >> 7;
        const int rem = p & 127;
        const int yo  = rem >> 5;
        const int z   = rem & 31;
        const int gx  = xx + xo - 1;
        const int gy  = yt * 2 + yo - 1;
        __bf16* dst = &lds[p * 64];
        if (gx >= 0 && gx < 32 && gy >= 0 && gy < 32) {
            const float* src = inb + ((gx * 32 + gy) * 32 + z);
            #pragma unroll
            for (int c = 0; c < CO; c += 2) {
                const float v0 = src[(size_t)c * R3];
                const float v1 = src[(size_t)(c + 1) * R3];
                dst[c]     = (__bf16)v0;
                dst[c + 1] = (__bf16)v1;
            }
        } else {
            #pragma unroll
            for (int c = 0; c < CO; ++c) dst[c] = (__bf16)0.0f;
        }
    }
    __syncthreads();

    // ---- compute: wave does 2 tiles (mo = tile>>2, pt = tile&3) ----
    for (int t = 0; t < 2; ++t) {
        const int tile   = wave * 2 + t;
        const int mo     = tile >> 2;
        const int pt     = tile & 3;
        const int yo_out = pt >> 1;
        const int zh     = pt & 1;
        v8f acc = {};
        for (int tap = 0; tap < 27; ++tap) {
            const int dz = tap % 3 - 1;
            const int dy = (tap / 3) % 3 - 1;
            const int dx = tap / 9 - 1;
            const int zi = zh * 16 + col + dz;
            const bool okz = (zi >= 0) && (zi < 32);
            const int posb = ((dx + 1) * 4 + (yo_out + dy + 1)) * 32;
            if (tap + 1 < 27)
                __builtin_prefetch((const void*)(wp + (((mo * 27 + tap + 1) * 2) << 9)), 0, 3);
            #pragma unroll
            for (int kc = 0; kc < 2; ++kc) {
                const v16bf a = *(const v16bf*)(wp +
                    ((((mo * 27 + tap) * 2 + kc) << 9) + (lane << 4)));
                v16bf bm = {};
                if (okz)
                    bm = *(const v16bf*)(&lds[(posb + zi) * 64 + kc * 32 + h * 16]);
                acc = __builtin_amdgcn_wmma_f32_16x16x32_bf16(false, a, false, bm,
                                                              (short)0, acc, false, false);
            }
        }
        const int gy = yt * 2 + yo_out;
        const size_t pos_out = (size_t)((xx * 32 + gy) * 32) + zh * 16 + col;
        #pragma unroll
        for (int r = 0; r < 8; ++r) {
            const int m = r + h * 8;
            out[(size_t)(b * CO + mo * 16 + m) * R3 + pos_out] = acc[r];
        }
    }
}

// ---------------------------------------------------------------------------
__global__ void k_zero(float* __restrict__ p, int n) {
    int i = blockIdx.x * 256 + threadIdx.x;
    if (i < n) p[i] = 0.0f;
}

__global__ void k_time_bias(const float* __restrict__ t_emb, const float* __restrict__ w_time,
                            const float* __restrict__ b_time, float* __restrict__ bias) {
    const int tid = threadIdx.x;            // 256 = B*CO
    const int b = tid >> 6, o = tid & 63;
    float s = b_time[o];
    for (int t = 0; t < TDIM; ++t) s += t_emb[b * TDIM + t] * w_time[o * TDIM + t];
    bias[b * CO + o] = s;
}

// GroupNorm stats per (b, group) over CPG channels x 32768 spatial
__global__ void k_gn_stats(const float* __restrict__ X, float* __restrict__ meanp,
                           float* __restrict__ rstdp) {
    __shared__ float s1[256], s2[256];
    const int bg = blockIdx.x;              // B*NGR = 32
    const int b = bg >> 3, g = bg & 7;
    const float* base = X + ((size_t)b * CO + g * CPG) * NP;
    float a = 0.0f, q = 0.0f;
    for (int i = threadIdx.x; i < CPG * NP; i += 256) {
        const float v = base[i];
        a += v; q += v * v;
    }
    s1[threadIdx.x] = a; s2[threadIdx.x] = q;
    __syncthreads();
    for (int o = 128; o > 0; o >>= 1) {
        if (threadIdx.x < o) {
            s1[threadIdx.x] += s1[threadIdx.x + o];
            s2[threadIdx.x] += s2[threadIdx.x + o];
        }
        __syncthreads();
    }
    if (threadIdx.x == 0) {
        const float cnt = (float)(CPG * NP);
        const float m = s1[0] / cnt;
        const float var = s2[0] / cnt - m * m;
        meanp[bg] = m;
        rstdp[bg] = rsqrtf(var + 1e-5f);
    }
}

// mode 0: silu(gn(x));  mode 1: + per-(b,c) bias;  mode 2: + elementwise extra
__global__ void k_gn_apply(const float* __restrict__ X, float* __restrict__ Y,
                           const float* __restrict__ meanp, const float* __restrict__ rstdp,
                           const float* __restrict__ gamma, const float* __restrict__ beta,
                           const float* __restrict__ extra, int mode) {
    const size_t i = (size_t)blockIdx.x * 256 + threadIdx.x;   // B*CO*NP
    const int c = (int)((i >> 15) & 63);
    const int b = (int)(i >> 21);
    const int bg = b * NGR + (c >> 3);
    const float x = X[i];
    const float xn = (x - meanp[bg]) * rstdp[bg] * gamma[c] + beta[c];
    float y = silu_f(xn);
    if (mode == 1)      y += extra[b * CO + c];
    else if (mode == 2) y += extra[i];
    Y[i] = y;
}

// scatter-add points onto voxel grid + occupancy counts
__global__ void k_voxelize(const float* __restrict__ X, const float* __restrict__ coords,
                           float* __restrict__ vox, float* __restrict__ cnt) {
    const int i = blockIdx.x * 256 + threadIdx.x;   // B*NP = 131072
    const int b = i >> 15, n = i & (NP - 1);
    const float* cc = coords + ((size_t)b * NP + n) * 3;
    const int ix = min(max((int)floorf(cc[0] * 31.0f), 0), 31);
    const int iy = min(max((int)floorf(cc[1] * 31.0f), 0), 31);
    const int iz = min(max((int)floorf(cc[2] * 31.0f), 0), 31);
    const int fi = (ix * 32 + iy) * 32 + iz;
    atomicAdd(&cnt[b * R3 + fi], 1.0f);
    for (int c = 0; c < CO; ++c)
        atomicAdd(&vox[(size_t)(b * CO + c) * R3 + fi],
                  X[(size_t)(b * CO + c) * NP + n]);
}

// vox = vox / max(cnt,1) + bias[b,c]
__global__ void k_vox_finalize(float* __restrict__ vox, const float* __restrict__ cnt,
                               const float* __restrict__ bias) {
    const size_t i = (size_t)blockIdx.x * 256 + threadIdx.x;   // B*CO*R3
    const int v = (int)(i & (R3 - 1));
    const int c = (int)((i >> 15) & 63);
    const int b = (int)(i >> 21);
    vox[i] = vox[i] / fmaxf(cnt[b * R3 + v], 1.0f) + bias[b * CO + c];
}

// trilinear gather back to points
__global__ void k_devox(const float* __restrict__ vox, const float* __restrict__ coords,
                        float* __restrict__ out) {
    const int i = blockIdx.x * 256 + threadIdx.x;   // B*NP
    const int b = i >> 15, n = i & (NP - 1);
    const float* cc = coords + ((size_t)b * NP + n) * 3;
    const float cx = cc[0] * 31.0f, cy = cc[1] * 31.0f, cz = cc[2] * 31.0f;
    const float fx = cx - floorf(cx), fy = cy - floorf(cy), fz = cz - floorf(cz);
    const int x0 = min(max((int)floorf(cx), 0), 31);
    const int y0 = min(max((int)floorf(cy), 0), 31);
    const int z0 = min(max((int)floorf(cz), 0), 31);
    const int x1 = min(x0 + 1, 31), y1 = min(y0 + 1, 31), z1 = min(z0 + 1, 31);
    const int xs[2] = {x0, x1}, ys[2] = {y0, y1}, zs[2] = {z0, z1};
    const float wx[2] = {1.0f - fx, fx}, wy[2] = {1.0f - fy, fy}, wz[2] = {1.0f - fz, fz};
    int fi[8]; float w8[8];
    int k = 0;
    #pragma unroll
    for (int dx = 0; dx < 2; ++dx)
        #pragma unroll
        for (int dy = 0; dy < 2; ++dy)
            #pragma unroll
            for (int dz = 0; dz < 2; ++dz) {
                fi[k] = (xs[dx] * 32 + ys[dy]) * 32 + zs[dz];
                w8[k] = wx[dx] * wy[dy] * wz[dz];
                ++k;
            }
    const float* vb = vox + (size_t)b * CO * R3;
    for (int c = 0; c < CO; ++c) {
        const float* vc = vb + (size_t)c * R3;
        float acc = 0.0f;
        #pragma unroll
        for (int j = 0; j < 8; ++j) acc += vc[fi[j]] * w8[j];
        out[(size_t)(b * CO + c) * NP + n] = acc;
    }
}

// ---------------------------------------------------------------------------
extern "C" void kernel_launch(void* const* d_in, const int* in_sizes, int n_in,
                              void* d_out, int out_size, void* d_ws, size_t ws_size,
                              hipStream_t stream) {
    const float* feats  = (const float*)d_in[0];
    const float* coords = (const float*)d_in[1];
    const float* t_emb  = (const float*)d_in[2];
    const float* w_in   = (const float*)d_in[3];
    const float* gn1_g  = (const float*)d_in[4];
    const float* gn1_b  = (const float*)d_in[5];
    const float* w_time = (const float*)d_in[6];
    const float* b_time = (const float*)d_in[7];
    const float* w_vox1 = (const float*)d_in[8];
    const float* gn2_g  = (const float*)d_in[9];
    const float* gn2_b  = (const float*)d_in[10];
    const float* w_vox2 = (const float*)d_in[11];
    const float* gn3_g  = (const float*)d_in[12];
    const float* gn3_b  = (const float*)d_in[13];
    const float* w_fuse = (const float*)d_in[14];
    const float* gn4_g  = (const float*)d_in[15];
    const float* gn4_b  = (const float*)d_in[16];
    const float* w_skip = (const float*)d_in[17];
    float* out = (float*)d_out;
    float* ws  = (float*)d_ws;

    const size_t S = (size_t)BB * CO * NP;      // 8,388,608 floats per big buffer
    float* bias  = ws;                          // 256
    float* meanb = ws + 256;                    // 32
    float* rstdb = ws + 512;                    // 32
    float* cnt   = ws + 768;                    // B*R3 = 131072
    float* xb    = cnt + (size_t)BB * R3;       // S
    float* vA    = xb + S;                      // S
    float* vB    = vA + S;                      // S
    __bf16* wp1  = (__bf16*)(vB + S);           // 110592 bf16
    __bf16* wp2  = wp1 + WPE;                   // 110592 bf16  (total ~101.6 MB)

    const dim3 blk(256);
    const int ew_grid = (int)(S / 256);         // 32768 blocks for elementwise over S

    // 0. pre-swizzle conv weights to bf16 WMMA fragment order
    k_prep_wconv<<<(WPE + 255) / 256, blk, 0, stream>>>(w_vox1, wp1);
    k_prep_wconv<<<(WPE + 255) / 256, blk, 0, stream>>>(w_vox2, wp2);
    // 1. point_in GEMM (WMMA, K=32)
    k_gemm_wmma<<<4096, blk, 0, stream>>>(w_in, feats, xb, CIN);
    // 2. GN1 stats, time bias
    k_gn_stats<<<32, blk, 0, stream>>>(xb, meanb, rstdb);
    k_time_bias<<<1, blk, 0, stream>>>(t_emb, w_time, b_time, bias);
    // 3. x_in = silu(gn1(x)) + bias  (in place)
    k_gn_apply<<<ew_grid, blk, 0, stream>>>(xb, xb, meanb, rstdb, gn1_g, gn1_b, bias, 1);
    // 4. voxelize (scatter-mean) + bias
    k_zero<<<(int)((S + 255) / 256), blk, 0, stream>>>(vA, (int)S);
    k_zero<<<(BB * R3 + 255) / 256, blk, 0, stream>>>(cnt, BB * R3);
    k_voxelize<<<(BB * NP) / 256, blk, 0, stream>>>(xb, coords, vA, cnt);
    k_vox_finalize<<<ew_grid, blk, 0, stream>>>(vA, cnt, bias);
    // 5. conv1 (WMMA + LDS halo) + GN2 + SiLU
    k_conv3d_wmma<<<2048, blk, 0, stream>>>(vA, wp1, vB);
    k_gn_stats<<<32, blk, 0, stream>>>(vB, meanb, rstdb);
    k_gn_apply<<<ew_grid, blk, 0, stream>>>(vB, vB, meanb, rstdb, gn2_g, gn2_b, nullptr, 0);
    // 6. conv2 + GN3 + SiLU
    k_conv3d_wmma<<<2048, blk, 0, stream>>>(vB, wp2, vA);
    k_gn_stats<<<32, blk, 0, stream>>>(vA, meanb, rstdb);
    k_gn_apply<<<ew_grid, blk, 0, stream>>>(vA, vA, meanb, rstdb, gn3_g, gn3_b, nullptr, 0);
    // 7. trilinear devoxelize -> xb
    k_devox<<<(BB * NP) / 256, blk, 0, stream>>>(vA, coords, xb);
    // 8. fuse GEMM (K=64) -> vB ; skip GEMM (K=32) -> vA
    k_gemm_wmma<<<4096, blk, 0, stream>>>(w_fuse, xb, vB, CO);
    k_gemm_wmma<<<4096, blk, 0, stream>>>(w_skip, feats, vA, CIN);
    // 9. out = silu(gn4(fuse)) + skip
    k_gn_stats<<<32, blk, 0, stream>>>(vB, meanb, rstdb);
    k_gn_apply<<<ew_grid, blk, 0, stream>>>(vB, out, meanb, rstdb, gn4_g, gn4_b, vA, 2);
}
